// DFlashAttentionMultiFn_66314295050597
// MI455X (gfx1250) — matrix operand order repaired and checked by reading, not compiled, not run on hardware
//
#include <hip/hip_runtime.h>
#include <hip/hip_bf16.h>
#include <stdint.h>

typedef __bf16 bf16_t;
typedef __attribute__((ext_vector_type(16))) __bf16 v16bf;
typedef __attribute__((ext_vector_type(8)))  float  v8f;

union FragBF { v16bf v; uint4 q[2]; };
union PK8  { bf16_t h[8]; uint4 u4; };
union PKD4 { bf16_t h[4]; uint2 u2; };

// ---------------- constants ----------------
#define HID   4096
#define NH    32
#define NKV   8
#define DH    128
#define LQ    64
#define TT    128       // CTX + L
#define WPOS  4096
#define ATOT  4224      // WPOS + TT = 132 * 32
#define NSPLIT 4
#define CHUNK 1056      // ATOT / NSPLIT = 33 * 32

// ======================================================================
// Generic GEMM: C[MxN](f32) = A[MxK](f32) * B[KxN](f32), bf16 WMMA, f32 acc
// block = 256 threads (8 waves), tile 32(M) x 64(N), K-step 64 per sync.
// Register double-buffering: next tile's global loads are issued while the
// current tile's fragments are read from LDS and fed to WMMA.
// Requires K % 64 == 0 (true for all uses here: K = 4096).
// ======================================================================
__global__ __launch_bounds__(256) void gemm_bf16_wmma(
    const float* __restrict__ A, const float* __restrict__ B,
    float* __restrict__ C, int M, int N, int K)
{
    __shared__ __align__(16) bf16_t As[32][72];   // [m][k], 64 k + 8 pad (144B rows)
    __shared__ __align__(16) bf16_t Bs[64][72];   // [n][k] transposed

    const int tid  = threadIdx.x;
    const int lane = tid & 31;
    const int wave = tid >> 5;       // 0..7
    const int wm   = wave & 1;
    const int wn   = wave >> 1;
    const int half = lane >> 4;
    const int l16  = lane & 15;

    const int bm0 = blockIdx.y * 32;
    const int bn0 = blockIdx.x * 64;

    // A cooperative load: 32 rows x 64 k, 8 consecutive k per thread
    const int ar = tid >> 3;             // 0..31
    const int ak = (tid & 7) * 8;        // 0,8,..,56
    // B cooperative load: 4 consecutive k x 4 consecutive n per thread
    const int bkp = (tid >> 4) * 4;      // 0,4,..,60
    const int bng = (tid & 15) * 4;      // 0,4,..,60

    const float* Arow  = A + (size_t)(bm0 + ar) * K + ak;
    const float* Bbase = B + bn0 + bng + (size_t)bkp * N;

    v8f acc = {};

    // ---- prologue: load first tile into registers ----
    float4 a0 = *(const float4*)(Arow);
    float4 a1 = *(const float4*)(Arow + 4);
    float4 b0 = *(const float4*)(Bbase);
    float4 b1 = *(const float4*)(Bbase + N);
    float4 b2 = *(const float4*)(Bbase + 2 * (size_t)N);
    float4 b3 = *(const float4*)(Bbase + 3 * (size_t)N);

    for (int k0 = 0; k0 < K; k0 += 64) {
        // ---- pack current tile into LDS ----
        PK8 pa;
        pa.h[0] = (bf16_t)a0.x; pa.h[1] = (bf16_t)a0.y;
        pa.h[2] = (bf16_t)a0.z; pa.h[3] = (bf16_t)a0.w;
        pa.h[4] = (bf16_t)a1.x; pa.h[5] = (bf16_t)a1.y;
        pa.h[6] = (bf16_t)a1.z; pa.h[7] = (bf16_t)a1.w;
        *(uint4*)&As[ar][ak] = pa.u4;

        PKD4 p0, p1, p2, p3;
        p0.h[0] = (bf16_t)b0.x; p0.h[1] = (bf16_t)b1.x;
        p0.h[2] = (bf16_t)b2.x; p0.h[3] = (bf16_t)b3.x;
        p1.h[0] = (bf16_t)b0.y; p1.h[1] = (bf16_t)b1.y;
        p1.h[2] = (bf16_t)b2.y; p1.h[3] = (bf16_t)b3.y;
        p2.h[0] = (bf16_t)b0.z; p2.h[1] = (bf16_t)b1.z;
        p2.h[2] = (bf16_t)b2.z; p2.h[3] = (bf16_t)b3.z;
        p3.h[0] = (bf16_t)b0.w; p3.h[1] = (bf16_t)b1.w;
        p3.h[2] = (bf16_t)b2.w; p3.h[3] = (bf16_t)b3.w;
        *(uint2*)&Bs[bng + 0][bkp] = p0.u2;
        *(uint2*)&Bs[bng + 1][bkp] = p1.u2;
        *(uint2*)&Bs[bng + 2][bkp] = p2.u2;
        *(uint2*)&Bs[bng + 3][bkp] = p3.u2;
        __syncthreads();

        // ---- issue next tile's global loads (overlap with WMMA below) ----
        if (k0 + 64 < K) {
            const float* An = Arow + k0 + 64;
            const float* Bn = Bbase + (size_t)(k0 + 64) * N;
            a0 = *(const float4*)(An);
            a1 = *(const float4*)(An + 4);
            b0 = *(const float4*)(Bn);
            b1 = *(const float4*)(Bn + N);
            b2 = *(const float4*)(Bn + 2 * (size_t)N);
            b3 = *(const float4*)(Bn + 3 * (size_t)N);
        }

        // ---- two 16x16x32 WMMAs from LDS fragments ----
#pragma unroll
        for (int s = 0; s < 2; ++s) {
            FragBF af, bfr;
            const bf16_t* ap = &As[wm * 16 + l16][s * 32 + half * 8];
            af.q[0] = *(const uint4*)(ap);
            af.q[1] = *(const uint4*)(ap + 16);
            const bf16_t* bq = &Bs[wn * 16 + l16][s * 32 + half * 8];
            bfr.q[0] = *(const uint4*)(bq);
            bfr.q[1] = *(const uint4*)(bq + 16);
            acc = __builtin_amdgcn_wmma_f32_16x16x32_bf16(
                false, af.v, false, bfr.v, (short)0, acc, false, false);
        }
        __syncthreads();
    }

    const int cm = bm0 + wm * 16 + half * 8;
    const int cn = bn0 + wn * 16 + l16;
#pragma unroll
    for (int r = 0; r < 8; ++r)
        C[(size_t)(cm + r) * N + cn] = acc[r];
}

// ======================================================================
// RMSNorm + RoPE for q -> bf16, pre-scaled by D^-0.5
// ======================================================================
__global__ __launch_bounds__(128) void rope_q_kernel(
    const float* __restrict__ qraw, const float* __restrict__ cosq,
    const float* __restrict__ sinq, const float* __restrict__ w,
    bf16_t* __restrict__ qbf)
{
    const int l = blockIdx.x, h = blockIdx.y, d = threadIdx.x;
    __shared__ float xs[128];
    __shared__ float red[4];

    float v = qraw[l * HID + h * DH + d];
    float ss = v * v;
#pragma unroll
    for (int m = 16; m; m >>= 1) ss += __shfl_xor(ss, m, 32);
    if ((threadIdx.x & 31) == 0) red[threadIdx.x >> 5] = ss;
    __syncthreads();
    float tot = red[0] + red[1] + red[2] + red[3];
    float rs = rsqrtf(tot * (1.0f / 128.0f) + 1e-6f);
    float xn = v * rs * w[d];
    xs[d] = xn;
    __syncthreads();
    float c = cosq[l * 64 + (d & 63)];
    float s = sinq[l * 64 + (d & 63)];
    float out = (d < 64) ? (xn * c - xs[d + 64] * s)
                         : (xn * c + xs[d - 64] * s);
    out *= 0.08838834764831845f;   // 1/sqrt(128)
    qbf[((size_t)h * LQ + l) * DH + d] = (bf16_t)out;
}

// ======================================================================
// RMSNorm + RoPE for k -> f32 output + bf16 cache tail
// ======================================================================
__global__ __launch_bounds__(128) void rope_k_kernel(
    const float* __restrict__ kraw, const float* __restrict__ cosk,
    const float* __restrict__ sink, const float* __restrict__ w,
    float* __restrict__ k_out, bf16_t* __restrict__ Kb)
{
    const int t = blockIdx.x, g = blockIdx.y, d = threadIdx.x;
    __shared__ float xs[128];
    __shared__ float red[4];

    float v = kraw[t * (NKV * DH) + g * DH + d];
    float ss = v * v;
#pragma unroll
    for (int m = 16; m; m >>= 1) ss += __shfl_xor(ss, m, 32);
    if ((threadIdx.x & 31) == 0) red[threadIdx.x >> 5] = ss;
    __syncthreads();
    float tot = red[0] + red[1] + red[2] + red[3];
    float rs = rsqrtf(tot * (1.0f / 128.0f) + 1e-6f);
    float xn = v * rs * w[d];
    xs[d] = xn;
    __syncthreads();
    float c = cosk[t * 64 + (d & 63)];
    float s = sink[t * 64 + (d & 63)];
    float out = (d < 64) ? (xn * c - xs[d + 64] * s)
                         : (xn * c + xs[d - 64] * s);
    k_out[((size_t)g * TT + t) * DH + d] = out;
    Kb[((size_t)g * ATOT + WPOS + t) * DH + d] = (bf16_t)out;
}

// v: vraw[128][1024] -> v_out[8][128][128](f32) and Vt[g][d][4096+t](bf16)
__global__ void scatter_v(const float* __restrict__ vraw,
                          float* __restrict__ v_out, bf16_t* __restrict__ Vt)
{
    int idx = blockIdx.x * blockDim.x + threadIdx.x;   // 0..131071
    int d = idx & 127, g = (idx >> 7) & 7, t = idx >> 10;
    float val = vraw[t * (NKV * DH) + g * DH + d];
    v_out[((size_t)g * TT + t) * DH + d] = val;
    Vt[((size_t)g * DH + d) * ATOT + WPOS + t] = (bf16_t)val;
}

// cache K -> Kb (copy + convert, coalesced)
__global__ void pack_K(const float* __restrict__ cK, bf16_t* __restrict__ Kb)
{
    size_t idx = (size_t)blockIdx.x * 256 + threadIdx.x;  // 8*4096*128
    int d = idx & 127;
    size_t gs = idx >> 7;
    int s = gs & 4095, g = gs >> 12;
    Kb[((size_t)g * ATOT + s) * DH + d] =
        (bf16_t)cK[((size_t)g * 8192 + s) * DH + d];
}

// cache V -> Vt transposed, LDS-tiled
__global__ __launch_bounds__(256) void pack_Vt(const float* __restrict__ cV,
                                               bf16_t* __restrict__ Vt)
{
    __shared__ float tile[32][33];
    int s0 = blockIdx.x * 32, d0 = blockIdx.y * 32, g = blockIdx.z;
    int tx = threadIdx.x & 31, ty = threadIdx.x >> 5;   // 32 x 8
#pragma unroll
    for (int i = 0; i < 32; i += 8)
        tile[ty + i][tx] = cV[((size_t)g * 8192 + s0 + ty + i) * DH + d0 + tx];
    __syncthreads();
#pragma unroll
    for (int i = 0; i < 32; i += 8)
        Vt[((size_t)g * DH + d0 + ty + i) * ATOT + s0 + tx] =
            (bf16_t)tile[tx][ty + i];
}

// ======================================================================
// Split-K flash attention: grid (32 heads x 4 key-splits), 128 thr (4 waves
// x 16 rows). Each block streams 1056 keys in 32-key WMMA blocks and writes
// unnormalized partials (O, m, l). global_prefetch hides K/V stream latency.
// ======================================================================
__global__ __launch_bounds__(128) void attn_kernel(
    const bf16_t* __restrict__ qbf,   // [32][64][128]
    const bf16_t* __restrict__ Kb,    // [8][4224][128]
    const bf16_t* __restrict__ Vt,    // [8][128][4224]
    float* __restrict__ Opart,        // [32][4][64][128]
    float* __restrict__ Mpart,        // [32][4][64]
    float* __restrict__ Lpart)        // [32][4][64]
{
    const int h  = blockIdx.x;
    const int sp = blockIdx.y;
    const int g  = h >> 2;
    const int wave = threadIdx.x >> 5;
    const int lane = threadIdx.x & 31;
    const int half = lane >> 4;
    const int l16  = lane & 15;

    __shared__ __align__(16) bf16_t Ps[4][16][40];

    // preload Q A-fragments (4 k-steps of 32 over D=128)
    FragBF qf[4];
    const bf16_t* qbase = qbf + ((size_t)h * LQ + wave * 16 + l16) * DH;
#pragma unroll
    for (int s = 0; s < 4; ++s) {
        const bf16_t* p = qbase + s * 32 + half * 8;
        qf[s].q[0] = *(const uint4*)(p);
        qf[s].q[1] = *(const uint4*)(p + 16);
    }

    v8f accO[8];
#pragma unroll
    for (int i = 0; i < 8; ++i) accO[i] = (v8f){};
    float mrow[8], lrow[8];
#pragma unroll
    for (int r = 0; r < 8; ++r) { mrow[r] = -1e30f; lrow[r] = 0.f; }

    const bf16_t* Kg = Kb + (size_t)g * ATOT * DH;
    const bf16_t* Vg = Vt + (size_t)g * DH * ATOT;

    const int kstart = sp * CHUNK;
    const int kend   = kstart + CHUNK;

    for (int kb = kstart; kb < kend; kb += 32) {
        // speculative prefetch of next 32-key block (dropped if OOB)
        __builtin_prefetch(Kg + (size_t)(kb + 32 + lane) * DH, 0, 3);
#pragma unroll
        for (int j = 0; j < 4; ++j)
            __builtin_prefetch(Vg + (size_t)(j * 32 + lane) * ATOT + kb + 32,
                               0, 3);

        // ---- scores: two 16x16 tiles over 32 keys ----
        v8f sc[2];
        sc[0] = (v8f){}; sc[1] = (v8f){};
#pragma unroll
        for (int t = 0; t < 2; ++t) {
            const bf16_t* krow =
                Kg + (size_t)(kb + t * 16 + l16) * DH + half * 8;
#pragma unroll
            for (int s = 0; s < 4; ++s) {
                FragBF kf;
                kf.q[0] = *(const uint4*)(krow + s * 32);
                kf.q[1] = *(const uint4*)(krow + s * 32 + 16);
                sc[t] = __builtin_amdgcn_wmma_f32_16x16x32_bf16(
                    false, qf[s].v, false, kf.v, (short)0, sc[t], false, false);
            }
        }

        // ---- online softmax (rows live across 16-lane halves) ----
        float p0[8], p1[8];
#pragma unroll
        for (int r = 0; r < 8; ++r) {
            float v0 = sc[0][r], v1 = sc[1][r];
            float vm = fmaxf(v0, v1);
#pragma unroll
            for (int m = 8; m; m >>= 1) vm = fmaxf(vm, __shfl_xor(vm, m, 32));
            float newm  = fmaxf(mrow[r], vm);
            float alpha = __expf(mrow[r] - newm);
            mrow[r] = newm;
            float e0 = __expf(v0 - newm), e1 = __expf(v1 - newm);
            p0[r] = e0; p1[r] = e1;
            float rsum = e0 + e1;
#pragma unroll
            for (int m = 8; m; m >>= 1) rsum += __shfl_xor(rsum, m, 32);
            lrow[r] = lrow[r] * alpha + rsum;
#pragma unroll
            for (int dt = 0; dt < 8; ++dt) accO[dt][r] *= alpha;
        }

        // ---- stage P through LDS: C-layout -> A-fragment layout ----
#pragma unroll
        for (int r = 0; r < 8; ++r) {
            Ps[wave][half * 8 + r][l16]      = (bf16_t)p0[r];
            Ps[wave][half * 8 + r][16 + l16] = (bf16_t)p1[r];
        }
        asm volatile("s_wait_dscnt 0" ::: "memory");   // in-wave LDS RAW

        FragBF pf;
        const bf16_t* pp = &Ps[wave][l16][half * 8];
        pf.q[0] = *(const uint4*)(pp);
        pf.q[1] = *(const uint4*)(pp + 16);

        // ---- O += P @ V (8 d-tiles, one 32-key WMMA each) ----
#pragma unroll
        for (int dt = 0; dt < 8; ++dt) {
            const bf16_t* vrow =
                Vg + (size_t)(dt * 16 + l16) * ATOT + kb + half * 8;
            FragBF vf;
            vf.q[0] = *(const uint4*)(vrow);
            vf.q[1] = *(const uint4*)(vrow + 16);
            accO[dt] = __builtin_amdgcn_wmma_f32_16x16x32_bf16(
                false, pf.v, false, vf.v, (short)0, accO[dt], false, false);
        }
    }

    // ---- write unnormalized partials ----
    const size_t base = ((size_t)h * NSPLIT + sp) * LQ;
#pragma unroll
    for (int r = 0; r < 8; ++r) {
        int row = wave * 16 + half * 8 + r;
        float* op = Opart + (base + row) * DH;
#pragma unroll
        for (int dt = 0; dt < 8; ++dt)
            op[dt * 16 + l16] = accO[dt][r];
        if (l16 == 0) {
            Mpart[base + row] = mrow[r];
            Lpart[base + row] = lrow[r];
        }
    }
}

// combine split partials -> attn[l][h*128 + d]
__global__ __launch_bounds__(128) void attn_combine(
    const float* __restrict__ Opart, const float* __restrict__ Mpart,
    const float* __restrict__ Lpart, float* __restrict__ attn)
{
    const int l = blockIdx.x;   // 0..63
    const int h = blockIdx.y;   // 0..31
    const int d = threadIdx.x;  // 0..127

    float M = -1e30f;
#pragma unroll
    for (int s = 0; s < NSPLIT; ++s)
        M = fmaxf(M, Mpart[((size_t)h * NSPLIT + s) * LQ + l]);
    float L = 0.f, O = 0.f;
#pragma unroll
    for (int s = 0; s < NSPLIT; ++s) {
        size_t idx = ((size_t)h * NSPLIT + s) * LQ + l;
        float wgt = __expf(Mpart[idx] - M);
        L += wgt * Lpart[idx];
        O += wgt * Opart[idx * DH + d];
    }
    attn[(size_t)l * HID + h * DH + d] = O / L;
}

// ======================================================================
// launcher
// ======================================================================
extern "C" void kernel_launch(void* const* d_in, const int* in_sizes, int n_in,
                              void* d_out, int out_size, void* d_ws, size_t ws_size,
                              hipStream_t stream)
{
    const float* x      = (const float*)d_in[0];
    const float* x_ctx  = (const float*)d_in[1];
    const float* cos_q  = (const float*)d_in[2];
    const float* sin_q  = (const float*)d_in[3];
    const float* cos_k  = (const float*)d_in[4];
    const float* sin_k  = (const float*)d_in[5];
    const float* cacheK = (const float*)d_in[6];
    const float* cacheV = (const float*)d_in[7];
    const float* Wq     = (const float*)d_in[9];
    const float* Wk     = (const float*)d_in[10];
    const float* Wv     = (const float*)d_in[11];
    const float* Wo     = (const float*)d_in[12];
    const float* qnw    = (const float*)d_in[13];
    const float* knw    = (const float*)d_in[14];

    char* ws = (char*)d_ws;
    float*  Cc    = (float*)(ws + 0);           // 128x4096 f32      (2 MB)
    float*  qraw  = (float*)(ws + 2097152);     // 64x4096 f32       (1 MB)
    float*  kraw  = (float*)(ws + 3145728);     // 128x1024 f32      (.5 MB)
    float*  vraw  = (float*)(ws + 3670016);     // 128x1024 f32      (.5 MB)
    float*  attn  = (float*)(ws + 4194304);     // 64x4096 f32       (1 MB)
    bf16_t* qb    = (bf16_t*)(ws + 5242880);    // 32x64x128 bf16    (.5 MB)
    bf16_t* Kb    = (bf16_t*)(ws + 5767168);    // 8x4224x128 bf16   (8.65 MB)
    bf16_t* Vt    = (bf16_t*)(ws + 14417920);   // 8x128x4224 bf16   (8.65 MB)
    float*  Opart = (float*)(ws + 23068672);    // 32x4x64x128 f32   (4 MB)
    float*  Mpart = (float*)(ws + 27262976);    // 32x4x64 f32
    float*  Lpart = (float*)(ws + 27295744);    // 32x4x64 f32

    float* out0 = (float*)d_out;               // 64x4096
    float* kout = out0 + 262144;               // 8x128x128
    float* vout = kout + 131072;               // 8x128x128

    // c = concat(x_ctx, x)
    hipMemcpyAsync(Cc,            x_ctx, (size_t)LQ * HID * 4,
                   hipMemcpyDeviceToDevice, stream);
    hipMemcpyAsync(Cc + LQ * HID, x,     (size_t)LQ * HID * 4,
                   hipMemcpyDeviceToDevice, stream);

    // projections
    gemm_bf16_wmma<<<dim3(HID / 64, LQ / 32), 256, 0, stream>>>(
        x, Wq, qraw, LQ, HID, HID);
    gemm_bf16_wmma<<<dim3((NKV * DH) / 64, TT / 32), 256, 0, stream>>>(
        Cc, Wk, kraw, TT, NKV * DH, HID);
    gemm_bf16_wmma<<<dim3((NKV * DH) / 64, TT / 32), 256, 0, stream>>>(
        Cc, Wv, vraw, TT, NKV * DH, HID);

    // KV cache packing to bf16 (K row-major, V transposed)
    pack_K<<<(NKV * WPOS * DH) / 256, 256, 0, stream>>>(cacheK, Kb);
    pack_Vt<<<dim3(WPOS / 32, DH / 32, NKV), 256, 0, stream>>>(cacheV, Vt);

    // norm + rope + layout
    rope_q_kernel<<<dim3(LQ, NH), 128, 0, stream>>>(qraw, cos_q, sin_q, qnw, qb);
    rope_k_kernel<<<dim3(TT, NKV), 128, 0, stream>>>(kraw, cos_k, sin_k, knw,
                                                     kout, Kb);
    scatter_v<<<(TT * NKV * DH) / 256, 256, 0, stream>>>(vraw, vout, Vt);

    // split-K flash attention + combine
    attn_kernel<<<dim3(NH, NSPLIT), 128, 0, stream>>>(qb, Kb, Vt,
                                                      Opart, Mpart, Lpart);
    attn_combine<<<dim3(LQ, NH), 128, 0, stream>>>(Opart, Mpart, Lpart, attn);

    // output projection
    gemm_bf16_wmma<<<dim3(HID / 64, LQ / 32), 256, 0, stream>>>(
        attn, Wo, out0, LQ, HID, HID);
}
